// encoder_decoder_GNN_90452011254230
// MI455X (gfx1250) — matrix operand order repaired and checked by reading, not compiled, not run on hardware
//
#include <hip/hip_runtime.h>
#include <hip/hip_bf16.h>

// ---------------------------------------------------------------------------
// Types for WMMA (gfx1250, wave32)
// ---------------------------------------------------------------------------
typedef __attribute__((ext_vector_type(16))) __bf16          v16bf;
typedef __attribute__((ext_vector_type(16))) unsigned short  v16u;
typedef __attribute__((ext_vector_type(8)))  float           v8f;

struct U32x8 { uint4 a, b; };   // 32 bytes, for bit_cast into v16u

// float -> bf16 bits, round-to-nearest-even
static __device__ __forceinline__ unsigned short f2bf(float x) {
  unsigned u = __float_as_uint(x);
  unsigned r = (u + 0x7FFFu + ((u >> 16) & 1u)) >> 16;
  return (unsigned short)r;
}

// pack 4 floats (float4) into au elements [e0..e0+3]
static __device__ __forceinline__ void pack4(v16u& au, int e0, float4 v) {
  au[e0 + 0] = f2bf(v.x);
  au[e0 + 1] = f2bf(v.y);
  au[e0 + 2] = f2bf(v.z);
  au[e0 + 3] = f2bf(v.w);
}

// ---------------------------------------------------------------------------
// WMMA GEMM:  Out(N x F) = A(N x Kd) @ W(Kd x F) [+ bias | += existing Out]
// One wave computes CT adjacent 16x16 f32 tiles from ONE shared A fragment
// (CT=6 covers a full 16x96 slab -> A is read exactly once per GEMM).
// W is staged TRANSPOSED into LDS as bf16, K zero-padded to KP (mult of 32),
// so all fragment loads are unguarded vector loads (b128).
// ---------------------------------------------------------------------------
template <int CT>
__global__ void __launch_bounds__(256) wmma_gemm_kernel(
    const float* __restrict__ A, const float* __restrict__ W,
    const float* __restrict__ bias, float* __restrict__ Out,
    int N, int Kd, int F, int accumulate) {
  __shared__ unsigned short Wlds[96 * 96];  // transposed: [n*KP + k], <= 18 KB

  const int KP = (Kd + 31) & ~31;           // zero-padded K stride
  const int sTotal = F * KP;
  for (int idx = threadIdx.x; idx < sTotal; idx += blockDim.x) {
    int nn = idx / KP;
    int k  = idx - nn * KP;
    Wlds[idx] = (k < Kd) ? f2bf(W[k * F + nn]) : (unsigned short)0;
  }
  __syncthreads();

  const int wave      = threadIdx.x >> 5;
  const int lane      = threadIdx.x & 31;
  const int colGroups = F / (16 * CT);
  const int rowTiles  = (N + 15) >> 4;
  const int tile      = blockIdx.x * (blockDim.x >> 5) + wave;
  if (tile >= rowTiles * colGroups) return;

  const int rowBase = (tile / colGroups) << 4;
  const int colBase = (tile % colGroups) * (16 * CT);
  const int half    = lane >> 4;       // 0: lanes 0-15, 1: lanes 16-31
  const int l16     = lane & 15;
  const int n       = colBase + l16;   // first output column for this lane
  int aRow = rowBase + l16;
  if (aRow >= N) aRow = N - 1;         // clamp: garbage rows never stored

  const bool fullTile = (rowBase + 16 <= N);

  // C/D layout: lane L, vgpr j -> row = rowBase + j + 8*half, col = l16
  v8f c[CT];
  if (accumulate) {
#pragma unroll
    for (int t = 0; t < CT; ++t) {
      if (fullTile) {
#pragma unroll
        for (int j = 0; j < 8; ++j)
          c[t][j] = Out[(size_t)(rowBase + j + (half << 3)) * F + n + 16 * t];
      } else {
#pragma unroll
        for (int j = 0; j < 8; ++j) {
          int r = rowBase + j + (half << 3);
          c[t][j] = (r < N) ? Out[(size_t)r * F + n + 16 * t] : 0.0f;
        }
      }
    }
  } else {
#pragma unroll
    for (int t = 0; t < CT; ++t) {
      float bv = bias ? bias[n + 16 * t] : 0.0f;
#pragma unroll
      for (int j = 0; j < 8; ++j) c[t][j] = bv;
    }
  }

  const float* Arow = A + (size_t)aRow * Kd;
  const int kFull = Kd & ~31;

  for (int kk = 0; kk < kFull; kk += 32) {
    // A fragment (16-bit A 16x32): elems 0-7  -> K = kk + 8*half + e
    //                              elems 8-15 -> K = kk + 16 + 8*half + e
    const float4* ap = (const float4*)(Arow + kk + (half << 3));
    v16u au;
    pack4(au, 0, ap[0]);
    pack4(au, 4, ap[1]);
    pack4(au, 8, ap[4]);   // +16 floats
    pack4(au, 12, ap[5]);
    v16bf a = __builtin_bit_cast(v16bf, au);
#pragma unroll
    for (int t = 0; t < CT; ++t) {
      // B fragment (16-bit B 32x16): elems 0-7 -> K = kk + 8*half + e,
      // elems 8-15 -> K = kk + 16 + 8*half + e, col n+16t
      const uint4* bp =
          (const uint4*)(Wlds + (n + 16 * t) * KP + kk + (half << 3));
      U32x8 braw{bp[0], bp[2]};
      v16bf b = __builtin_bit_cast(v16bf, __builtin_bit_cast(v16u, braw));
      c[t] = __builtin_amdgcn_wmma_f32_16x16x32_bf16(false, a, false, b,
                                                     (short)0, c[t], false, false);
    }
  }

  if (Kd & 31) {  // remainder == 16 (Kd = 16 layers): upper half of A is zero
    const float4* ap = (const float4*)(Arow + kFull + (half << 3));
    v16u au;
    pack4(au, 0, ap[0]);
    pack4(au, 4, ap[1]);
#pragma unroll
    for (int e = 8; e < 16; ++e) au[e] = 0;
    v16bf a = __builtin_bit_cast(v16bf, au);
#pragma unroll
    for (int t = 0; t < CT; ++t) {
      const uint4* bp =
          (const uint4*)(Wlds + (n + 16 * t) * KP + kFull + (half << 3));
      U32x8 braw{bp[0], bp[2]};   // padded zeros cover K >= Kd
      v16bf b = __builtin_bit_cast(v16bf, __builtin_bit_cast(v16u, braw));
      c[t] = __builtin_amdgcn_wmma_f32_16x16x32_bf16(false, a, false, b,
                                                     (short)0, c[t], false, false);
    }
  }

#pragma unroll
  for (int t = 0; t < CT; ++t) {
    if (fullTile) {
#pragma unroll
      for (int j = 0; j < 8; ++j)
        Out[(size_t)(rowBase + j + (half << 3)) * F + n + 16 * t] = c[t][j];
    } else {
#pragma unroll
      for (int j = 0; j < 8; ++j) {
        int r = rowBase + j + (half << 3);
        if (r < N) Out[(size_t)r * F + n + 16 * t] = c[t][j];
      }
    }
  }
}

// ---------------------------------------------------------------------------
// Graph preprocessing
// ---------------------------------------------------------------------------
__global__ void degree_kernel(const int* __restrict__ row, float* __restrict__ deg, int E) {
  int e = blockIdx.x * blockDim.x + threadIdx.x;
  if (e < E) atomicAdd(&deg[row[e]], 1.0f);
}

__global__ void dinv_kernel(float* __restrict__ deg, int N) {
  int i = blockIdx.x * blockDim.x + threadIdx.x;
  if (i < N) {
    float d = deg[i];
    deg[i] = (d > 0.0f) ? rsqrtf(d) : 0.0f;
  }
}

__global__ void enorm_kernel(const int* __restrict__ row, const int* __restrict__ col,
                             const float* __restrict__ dinv, float* __restrict__ enorm, int E) {
  int e = blockIdx.x * blockDim.x + threadIdx.x;
  if (e < E) enorm[e] = -(dinv[row[e]] * dinv[col[e]]);
}

// ---------------------------------------------------------------------------
// prop: Out[col[e], :] += scale * enorm[e] * t[row[e], :]  (F=96, 4 feat/thread)
// With Out pre-initialized to -Tx0 and scale=2 this directly yields
// the Chebyshev recurrence  Tx2 = 2*L_hat@Tx1 - Tx0.
// ---------------------------------------------------------------------------
__global__ void prop_kernel(const float* __restrict__ T, const int* __restrict__ row,
                            const int* __restrict__ col, const float* __restrict__ enorm,
                            float* __restrict__ Out, int E, float scale) {
  const int CH = 96 / 4;  // 24 threads per edge
  int tid = blockIdx.x * blockDim.x + threadIdx.x;
  if (tid >= E * CH) return;
  int e  = tid / CH;
  int fo = (tid - e * CH) * 4;
  int r = row[e], cidx = col[e];
  float w = scale * enorm[e];
  float4 v = *(const float4*)(T + (size_t)r * 96 + fo);
  float* o = Out + (size_t)cidx * 96 + fo;
  atomicAdd(o + 0, w * v.x);
  atomicAdd(o + 1, w * v.y);
  atomicAdd(o + 2, w * v.z);
  atomicAdd(o + 3, w * v.w);
}

// O = -S (vectorized negating copy; scatter-target init for Chebyshev step)
__global__ void negcopy_kernel(float4* __restrict__ O, const float4* __restrict__ S, int n4) {
  int i = blockIdx.x * blockDim.x + threadIdx.x;
  if (i < n4) {
    float4 s = S[i];
    s.x = -s.x; s.y = -s.y; s.z = -s.z; s.w = -s.w;
    O[i] = s;
  }
}

// ---------------------------------------------------------------------------
// BatchNorm (training-mode, biased stats over node dim) + ReLU
// ---------------------------------------------------------------------------
__global__ void __launch_bounds__(256) col_stats_kernel(
    const float* __restrict__ X, int N, int F,
    const float* __restrict__ g, const float* __restrict__ be,
    float* __restrict__ scale, float* __restrict__ shift) {
  int f = blockIdx.x;
  float s = 0.0f, s2 = 0.0f;
  for (int i = threadIdx.x; i < N; i += blockDim.x) {
    float v = X[(size_t)i * F + f];
    s += v;
    s2 += v * v;
  }
  __shared__ float sh1[256], sh2[256];
  sh1[threadIdx.x] = s;
  sh2[threadIdx.x] = s2;
  __syncthreads();
  for (int off = 128; off > 0; off >>= 1) {
    if (threadIdx.x < off) {
      sh1[threadIdx.x] += sh1[threadIdx.x + off];
      sh2[threadIdx.x] += sh2[threadIdx.x + off];
    }
    __syncthreads();
  }
  if (threadIdx.x == 0) {
    float mean = sh1[0] / (float)N;
    float var  = sh2[0] / (float)N - mean * mean;
    float sc   = g[f] * rsqrtf(var + 1e-5f);
    scale[f] = sc;
    shift[f] = be[f] - mean * sc;
  }
}

__global__ void bn_relu_kernel(float* __restrict__ X, int total, int F,
                               const float* __restrict__ scale,
                               const float* __restrict__ shift) {
  int i = blockIdx.x * blockDim.x + threadIdx.x;
  if (i < total) {
    int f = i % F;
    float v = X[i] * scale[f] + shift[f];
    X[i] = v > 0.0f ? v : 0.0f;
  }
}

// ---------------------------------------------------------------------------
// Host orchestration
// ---------------------------------------------------------------------------
extern "C" void kernel_launch(void* const* d_in, const int* in_sizes, int n_in,
                              void* d_out, int out_size, void* d_ws, size_t ws_size,
                              hipStream_t stream) {
  (void)n_in; (void)out_size; (void)ws_size;

  const float* x  = (const float*)d_in[0];
  const int*   ei = (const int*)d_in[1];
  const int N = in_sizes[0] / 16;
  const int E = in_sizes[1] / 2;
  const int* row = ei;
  const int* col = ei + E;

  // jax pytree flatten order: dict keys sorted -> dec, enc, gnn; inner sorted.
  const float* dec_b1  = (const float*)d_in[2];
  const float* dec_b2  = (const float*)d_in[3];
  const float* dec_b3  = (const float*)d_in[4];
  const float* dec_b4  = (const float*)d_in[5];
  const float* dec_be1 = (const float*)d_in[6];
  const float* dec_be2 = (const float*)d_in[7];
  const float* dec_g1  = (const float*)d_in[8];
  const float* dec_g2  = (const float*)d_in[9];
  const float* dec_w1  = (const float*)d_in[10];
  const float* dec_w2  = (const float*)d_in[11];
  const float* dec_w3  = (const float*)d_in[12];
  const float* dec_w4  = (const float*)d_in[13];
  const float* enc_b1  = (const float*)d_in[14];
  const float* enc_b2  = (const float*)d_in[15];
  const float* enc_b3  = (const float*)d_in[16];
  const float* enc_b4  = (const float*)d_in[17];
  const float* enc_be1 = (const float*)d_in[18];
  const float* enc_be2 = (const float*)d_in[19];
  const float* enc_g1  = (const float*)d_in[20];
  const float* enc_g2  = (const float*)d_in[21];
  const float* enc_w1  = (const float*)d_in[22];
  const float* enc_w2  = (const float*)d_in[23];
  const float* enc_w3  = (const float*)d_in[24];
  const float* enc_w4  = (const float*)d_in[25];
  const float* gnn_W[2]  = {(const float*)d_in[26], (const float*)d_in[30]};
  const float* gnn_b[2]  = {(const float*)d_in[27], (const float*)d_in[31]};
  const float* gnn_be[2] = {(const float*)d_in[28], (const float*)d_in[32]};
  const float* gnn_g[2]  = {(const float*)d_in[29], (const float*)d_in[33]};

  // ---- workspace carve-up --------------------------------------------------
  char* p = (char*)d_ws;
  auto carve = [&](size_t bytes) {
    char* q = p;
    p += (bytes + 255) & ~(size_t)255;
    return (void*)q;
  };
  float* bufA  = (float*)carve((size_t)N * 96 * sizeof(float));
  float* bufB  = (float*)carve((size_t)N * 96 * sizeof(float));
  float* bufC  = (float*)carve((size_t)N * 96 * sizeof(float));
  float* bufD  = (float*)carve((size_t)N * 96 * sizeof(float));
  float* enorm = (float*)carve((size_t)E * sizeof(float));
  float* deg   = (float*)carve((size_t)N * sizeof(float));
  float* scale = (float*)carve(96 * sizeof(float));
  float* shift = (float*)carve(96 * sizeof(float));

  // ---- launch helpers ------------------------------------------------------
  auto gemm = [&](const float* A, const float* W, const float* bias, float* O,
                  int K, int F, bool acc) {
    if (F % 96 == 0) {
      int tiles  = ((N + 15) / 16) * (F / 96);
      int blocks = (tiles + 7) / 8;  // 8 waves per 256-thread block
      wmma_gemm_kernel<6><<<blocks, 256, 0, stream>>>(A, W, bias, O, N, K, F,
                                                      acc ? 1 : 0);
    } else {
      int tiles  = ((N + 15) / 16) * (F / 16);
      int blocks = (tiles + 7) / 8;
      wmma_gemm_kernel<1><<<blocks, 256, 0, stream>>>(A, W, bias, O, N, K, F,
                                                      acc ? 1 : 0);
    }
  };
  auto bn = [&](float* X, int F, const float* g, const float* be) {
    col_stats_kernel<<<F, 256, 0, stream>>>(X, N, F, g, be, scale, shift);
    int tot = N * F;
    bn_relu_kernel<<<(tot + 255) / 256, 256, 0, stream>>>(X, tot, F, scale, shift);
  };
  auto prop = [&](const float* T, float* O, float sc) {
    int tot = E * 24;
    prop_kernel<<<(tot + 255) / 256, 256, 0, stream>>>(T, row, col, enorm, O, E, sc);
  };

  // ---- graph normalization -------------------------------------------------
  hipMemsetAsync(deg, 0, (size_t)N * sizeof(float), stream);
  degree_kernel<<<(E + 255) / 256, 256, 0, stream>>>(row, deg, E);
  dinv_kernel<<<(N + 255) / 256, 256, 0, stream>>>(deg, N);
  enorm_kernel<<<(E + 255) / 256, 256, 0, stream>>>(row, col, deg, enorm, E);

  // ---- encoder MLP ---------------------------------------------------------
  gemm(x, enc_w1, enc_b1, bufA, 16, 96, false);
  bn(bufA, 96, enc_g1, enc_be1);
  gemm(bufA, enc_w2, enc_b2, bufB, 96, 96, false);
  gemm(bufB, enc_w3, enc_b3, bufA, 96, 96, false);
  bn(bufA, 96, enc_g2, enc_be2);
  gemm(bufA, enc_w4, enc_b4, bufB, 96, 96, false);

  // ---- ChebConv GNN layers -------------------------------------------------
  float* pool[4] = {bufA, bufB, bufC, bufD};
  float* h = bufB;
  const int n4 = N * 96 / 4;
  for (int l = 0; l < 2; ++l) {
    float* others[3];
    int m = 0;
    for (int i = 0; i < 4; ++i)
      if (pool[i] != h) others[m++] = pool[i];
    float* acc   = others[0];
    float* Tx0   = h;
    float* Tx1   = others[1];
    float* spare = others[2];
    const float* W = gnn_W[l];

    gemm(Tx0, W, gnn_b[l], acc, 96, 96, false);           // k=0 (+bias)
    hipMemsetAsync(Tx1, 0, (size_t)N * 96 * sizeof(float), stream);
    prop(Tx0, Tx1, 1.0f);                                 // Tx1 = L_hat @ Tx0
    gemm(Tx1, W + 96 * 96, nullptr, acc, 96, 96, true);   // k=1
    for (int k = 2; k < 20; ++k) {
      negcopy_kernel<<<(n4 + 255) / 256, 256, 0, stream>>>(
          (float4*)spare, (const float4*)Tx0, n4);        // spare = -Tx0
      prop(Tx1, spare, 2.0f);                             // spare += 2*L_hat@Tx1
      gemm(spare, W + (size_t)k * 96 * 96, nullptr, acc, 96, 96, true);
      float* t = Tx0; Tx0 = Tx1; Tx1 = spare; spare = t;
    }
    bn(acc, 96, gnn_g[l], gnn_be[l]);
    h = acc;
  }

  // ---- decoder MLP ---------------------------------------------------------
  float* f1 = nullptr; float* f2 = nullptr;
  for (int i = 0; i < 4; ++i)
    if (pool[i] != h) { if (!f1) f1 = pool[i]; else if (!f2) f2 = pool[i]; }
  gemm(h, dec_w1, dec_b1, f1, 96, 96, false);
  bn(f1, 96, dec_g1, dec_be1);
  gemm(f1, dec_w2, dec_b2, f2, 96, 96, false);
  gemm(f2, dec_w3, dec_b3, f1, 96, 16, false);
  bn(f1, 16, dec_g2, dec_be2);
  gemm(f1, dec_w4, dec_b4, (float*)d_out, 16, 16, false);
}
